// GaussianImage_Cholesky_53420803227738
// MI455X (gfx1250) — matrix operand order repaired and checked by reading, not compiled, not run on hardware
//
#include <hip/hip_runtime.h>
#include <math.h>

// ---------------------------------------------------------------------------
// GaussianImage (Cholesky) renderer for MI455X / gfx1250, wave32 + WMMA.
//
// Compute-bound: 2*65536*1024 = 134M gaussian-pixel evals; params 64KB,
// output 1.5MB -> HBM irrelevant, VALU/trans pipe is the roofline.
//
// prep_kernel: per (t,n) fold time-poly + tanh/sigmoid + covariance inverse,
//   then expand sigma into a 6-term dot product vs per-pixel constants
//   (1, px, py, px^2, py^2, px*py) and premultiply by -log2(e) so the
//   render inner loop is exactly 5 FMA + 1 raw v_exp_f32 per eval.
//   Also emit opacity-premultiplied colors as a (3, N) plane.
//
// render_kernel: block = 8 waves = 128 pixels. Stage 32KB geometry + build
//   32KB of f16 B-tiles (exact WMMA B-matrix lane layout, zero cols 3..15)
//   in LDS once. Each wave owns 16 pixels (M) and marches K = gaussian in
//   chunks of 32: lanes compute w = exp2(dot) into the f16 A layout, B is
//   two unconditional ds_load_b128, and the color GEMM accumulates on the
//   matrix pipe via v_wmma_f32_16x16x32_f16 (co-executes with VALU/trans).
// ---------------------------------------------------------------------------

typedef __attribute__((ext_vector_type(16))) _Float16 v16h;
typedef __attribute__((ext_vector_type(8)))  float    v8f;

#define N_G   1024
#define W_IMG 256
#define H_IMG 256
#define T_FR  2
#define GPAR  8                      // floats per gaussian geometry row
#define NCHNK (N_G / 32)             // 32 K-chunks

#if __has_builtin(__builtin_amdgcn_exp2f)
#define FAST_EXP2(x) __builtin_amdgcn_exp2f(x)   // bare v_exp_f32, no ldexp fixup
#else
#define FAST_EXP2(x) exp2f(x)
#endif

__device__ __forceinline__ float sigmoidf_(float x) {
    return 1.0f / (1.0f + __expf(-x));
}

// ws layout: geom (T, N, 8) f32  |  pcol (T, 3, N) f32
__global__ __launch_bounds__(256)
void prep_kernel(const float* __restrict__ xyz,   // (3, N, 2)
                 const float* __restrict__ chol,  // (3, N, 3)
                 const float* __restrict__ opc,   // (3, N, 1)
                 const float* __restrict__ feat,  // (N, 3)
                 float* __restrict__ geom,        // (T, N, 8)
                 float* __restrict__ pcol)        // (T, 3, N)
{
    int i = blockIdx.x * blockDim.x + threadIdx.x;
    if (i >= T_FR * N_G) return;
    int t = i / N_G;
    int n = i - t * N_G;

    float tv  = (float)t / (float)(T_FR - 1);   // linspace(0,1,T)
    float tp1 = tv, tp2 = tv * tv;              // tp0 == 1

    // xy -> pixel-space means
    float sx = xyz[n*2+0] + tp1*xyz[N_G*2 + n*2+0] + tp2*xyz[2*N_G*2 + n*2+0];
    float sy = xyz[n*2+1] + tp1*xyz[N_G*2 + n*2+1] + tp2*xyz[2*N_G*2 + n*2+1];
    float mx = 0.5f * (tanhf(sx) + 1.0f) * (float)W_IMG;
    float my = 0.5f * (tanhf(sy) + 1.0f) * (float)H_IMG;

    // cholesky -> inverse covariance, halved for the quadratic form
    float l1 = chol[n*3+0] + tp1*chol[N_G*3+n*3+0] + tp2*chol[2*N_G*3+n*3+0] + 0.5f;
    float l2 = chol[n*3+1] + tp1*chol[N_G*3+n*3+1] + tp2*chol[2*N_G*3+n*3+1];
    float l3 = chol[n*3+2] + tp1*chol[N_G*3+n*3+2] + tp2*chol[2*N_G*3+n*3+2] + 0.5f;
    float s11 = l1*l1, s12 = l1*l2, s22 = l2*l2 + l3*l3;
    float det = fmaxf(s11*s22 - s12*s12, 1e-8f);
    float inv = 1.0f / det;
    float ha =  0.5f * s22 * inv;   // 0.5*a
    float b  = -s12 * inv;
    float hc =  0.5f * s11 * inv;   // 0.5*c

    // sigma(px,py) = c0 - qx*px - qy*py + ha*px^2 + hc*py^2 + b*px*py
    float c0 = ha*mx*mx + hc*my*my + b*mx*my;
    float qx = 2.0f*ha*mx + b*my;
    float qy = 2.0f*hc*my + b*mx;

    // premultiply by -log2(e): w = exp(-sigma) = exp2(g . pixvec)
    const float L = 1.4426950408889634f;
    float* o = geom + (size_t)i * GPAR;
    o[0] = -L * c0;   // * 1
    o[1] =  L * qx;   // * px
    o[2] =  L * qy;   // * py
    o[3] = -L * ha;   // * px^2
    o[4] = -L * hc;   // * py^2
    o[5] = -L * b;    // * px*py
    o[6] = 0.0f;
    o[7] = 0.0f;

    // opacity-premultiplied colors, (T, 3, N) plane for B-tile building
    float so   = opc[n] + tp1*opc[N_G + n] + tp2*opc[2*N_G + n];
    float opac = sigmoidf_(so);
    float* pc = pcol + (size_t)t * 3 * N_G;
    pc[0*N_G + n] = opac * sigmoidf_(feat[n*3+0]);
    pc[1*N_G + n] = opac * sigmoidf_(feat[n*3+1]);
    pc[2*N_G + n] = opac * sigmoidf_(feat[n*3+2]);
}

__global__ __launch_bounds__(256)
void render_kernel(const float* __restrict__ geomG,  // (T, N, 8)
                   const float* __restrict__ pcolG,  // (T, 3, N)
                   float* __restrict__ out)          // (T, 3, H, W)
{
    __shared__ __align__(32) float    geo[N_G * GPAR];       // 32 KB
    __shared__ __align__(32) unsigned bmt[NCHNK * 32 * 8];   // 32 KB f16-pair B tiles

    const int t   = blockIdx.y;
    const int tid = threadIdx.x;

    // --- Stage geometry (coalesced float4 copy) -------------------------
    {
        const float4* src = (const float4*)(geomG + (size_t)t * N_G * GPAR);
        float4* dst = (float4*)geo;
        #pragma unroll
        for (int i = 0; i < (N_G * GPAR / 4) / 256; ++i)
            dst[tid + i * 256] = src[tid + i * 256];
    }

    // --- Build f16 B tiles in the exact WMMA B-matrix layout ------------
    // dword idx = (chunk, lane, v): lane column N = lane&15 (channel, zero
    // for N>=3), rows K = chunk*32 + 8*(lane>>4) + 2v + 8*(v>>2) + {0, 1}.
    {
        const float* pc = pcolG + (size_t)t * 3 * N_G;
        for (int idx = tid; idx < NCHNK * 32 * 8; idx += 256) {
            int v     = idx & 7;
            int lane  = (idx >> 3) & 31;
            int chunk = idx >> 8;
            int n     = lane & 15;
            unsigned d = 0u;
            if (n < 3) {
                int k = chunk * 32 + ((lane >> 4) << 3) + 2 * v + ((v >> 2) << 3);
                union { _Float16 h[2]; unsigned u; } cv;
                cv.h[0] = (_Float16)pc[n * N_G + k];
                cv.h[1] = (_Float16)pc[n * N_G + k + 1];
                d = cv.u;
            }
            bmt[idx] = d;
        }
    }
    __syncthreads();

    const int wave = tid >> 5;
    const int lane = tid & 31;
    const int m    = lane & 15;   // A row (pixel) / C column (channel)
    const int hi   = lane >> 4;   // half-wave: K split for A, row split for C

    const int wb = blockIdx.x * 128 + wave * 16;      // wave's 16-pixel strip
    const int p  = wb + m;
    const float px   = (float)(p & (W_IMG - 1)) + 0.5f;
    const float py   = (float)(p >> 8) + 0.5f;
    const float px2  = px * px;
    const float py2  = py * py;
    const float pxpy = px * py;

    const v16h* bmat = (const v16h*)bmt;              // 32B per (chunk,lane)

    v8f acc;
    #pragma unroll
    for (int r = 0; r < 8; ++r) acc[r] = 0.0f;

    for (int c = 0; c < NCHNK; ++c) {
        const int k0 = c * 32;

        // A tile: w[16 pixels x 32 gaussians], f16 A layout:
        // lanes 0-15: K = {0..7, 16..23}; lanes 16-31: K = {8..15, 24..31};
        // VGPR v holds the packed f16 pair of its K range.
        v16h av;
        #pragma unroll
        for (int v = 0; v < 8; ++v) {
            const int kb = (v < 4) ? (k0 + 8*hi + 2*v)
                                   : (k0 + 16 + 8*hi + 2*(v - 4));
            #pragma unroll
            for (int h = 0; h < 2; ++h) {
                const float* g = &geo[(kb + h) * GPAR];  // half-wave broadcast
                float s = fmaf(g[1], px, g[0]);          // 5 FMA total
                s = fmaf(g[2], py,   s);
                s = fmaf(g[3], px2,  s);
                s = fmaf(g[4], py2,  s);
                s = fmaf(g[5], pxpy, s);
                av[2*v + h] = (_Float16)FAST_EXP2(s);    // bare v_exp_f32
            }
        }

        // B tile: two unconditional ds_load_b128 straight into WMMA sources.
        const v16h bv = bmat[c * 32 + lane];

        acc = __builtin_amdgcn_wmma_f32_16x16x32_f16(
                  false, av, false, bv, (short)0, acc, false, false);
    }

    // C/D layout: lane holds column N = m; VGPR r is row M = r + 8*hi.
    if (m < 3) {
        #pragma unroll
        for (int r = 0; r < 8; ++r) {
            const int M  = r + 8*hi;
            const int pp = wb + M;
            const int y  = pp >> 8;
            const int x  = pp & (W_IMG - 1);
            const float v = fminf(fmaxf(acc[r], 0.0f), 1.0f);
            out[(((size_t)t * 3 + m) * H_IMG + y) * W_IMG + x] = v;
        }
    }
}

extern "C" void kernel_launch(void* const* d_in, const int* in_sizes, int n_in,
                              void* d_out, int out_size, void* d_ws, size_t ws_size,
                              hipStream_t stream) {
    const float* xyz  = (const float*)d_in[0];  // (3, 1024, 2)
    const float* chol = (const float*)d_in[1];  // (3, 1024, 3)
    const float* opc  = (const float*)d_in[2];  // (3, 1024, 1)
    const float* feat = (const float*)d_in[3];  // (1024, 3)
    float* out = (float*)d_out;                 // (2, 3, 256, 256)

    float* geom = (float*)d_ws;                       // T*N*8 f32 = 64 KB
    float* pcol = geom + (size_t)T_FR * N_G * GPAR;   // T*3*N f32 = 24 KB

    prep_kernel<<<(T_FR * N_G + 255) / 256, 256, 0, stream>>>(
        xyz, chol, opc, feat, geom, pcol);

    dim3 grid(H_IMG * W_IMG / 128, T_FR);       // 8 waves * 16 pixels per block
    render_kernel<<<grid, 256, 0, stream>>>(geom, pcol, out);
}
